// NODE_DMD_9285719294021
// MI455X (gfx1250) — compile-verified, matrix operand order checked by reading, not run on hardware
//
#include <hip/hip_runtime.h>
#include <math.h>

// NODE-DMD fused implementation for gfx1250 (MI455X).
// Input order (setup_inputs insertion order, leaves w-then-b):
//  0 coords[4096,2] 1 y_prev[4096,2]
//  2..3  enc.l1.{w[4,128],b}   4..5  enc.l2.{w[128,128],b}
//  6..7  enc.pool.{w,b}        8..9  enc.mu.{w[128,32],b}
// 10..11 enc.logvar.{w,b}     12..13 enc.lam.{w,b}
// 14..15 ode.l1.{w[65,128],b} 16..17 ode.l2.{w,b} 18..19 ode.l3.{w[128,32],b}
// 20..21 mode.l1.{w[2,128],b} 22..23 mode.l2.{w,b} 24..25 mode.l3.{w[128,32],b}
// 26 t_prev 27 t_next
// Output (float32, concat): mu_u[8192] logvar_u[8192] mu[32] logvar[32] lam[32] W[131072]

#define LDA 136          // padded f16 row stride
#define NSTEPS 20
#define QNOISE 1e-3f
#define COVEPS 1e-6f

typedef __attribute__((ext_vector_type(16))) _Float16 v16h;
typedef __attribute__((ext_vector_type(8)))  _Float16 h8;
typedef __attribute__((ext_vector_type(8)))  float    v8f;

// A/B fragment per ISA 7.12.2: lane L -> row = (L&15); elements 0..7 hold
// K = kbase..kbase+7, elements 8..15 hold K = kbase+16..kbase+23, where
// kbase = kc*32 + (L>>4)*8.  Two 16-byte LDS loads per fragment (ds_load_b128).
__device__ __forceinline__ v16h load_frag(const _Float16* base, int row, int kbase) {
  const _Float16* p = base + row * LDA + kbase;
  h8 lo = *(const h8*)(p);
  h8 hi = *(const h8*)(p + 16);
  v16h a;
#pragma unroll
  for (int e = 0; e < 8; ++e) { a[e] = lo[e]; a[e + 8] = hi[e]; }
  return a;
}

__device__ __forceinline__ v8f wmma16(v16h a, v16h b, v8f c) {
  return __builtin_amdgcn_wmma_f32_16x16x32_f16(false, a, false, b, (short)0, c, false, false);
}

// Prefetch a float array into the cache hierarchy (global_prefetch_b8),
// one 128B cacheline per call, strided across the block.
__device__ __forceinline__ void prefetch_f32(const float* p, int nelem, int t) {
  for (int i = t * 32; i < nelem; i += 256 * 32) __builtin_prefetch(p + i, 0, 1);
}

// ---------------- Kernel 1: encoder hidden + column sums ----------------
__global__ __launch_bounds__(256) void enc_kernel(
    const float* __restrict__ coords, const float* __restrict__ yprev,
    const float* __restrict__ w1, const float* __restrict__ b1,
    const float* __restrict__ w2, const float* __restrict__ b2,
    float* __restrict__ partial) {
  __shared__ alignas(16) _Float16 Xh[128 * LDA];   // h1 activations, f16
  __shared__ alignas(16) _Float16 Wt[128 * LDA];   // enc.l2 W transposed [n][k], f16
  __shared__ float w1s[512], b1s[128], b2s[128], xs[128 * 4], wsum[8 * 128];

  const int t = threadIdx.x;
  const int n0 = blockIdx.x * 128;

  prefetch_f32(w2, 16384, t);                      // overlap L2 fill with staging below

  for (int i = t; i < 512; i += 256) w1s[i] = w1[i];
  for (int i = t; i < 128; i += 256) { b1s[i] = b1[i]; b2s[i] = b2[i]; }
  for (int i = t; i < 128; i += 256) {
    int n = n0 + i;
    xs[i * 4 + 0] = coords[2 * n];  xs[i * 4 + 1] = coords[2 * n + 1];
    xs[i * 4 + 2] = yprev[2 * n];   xs[i * 4 + 3] = yprev[2 * n + 1];
  }
  for (int i = t; i < 16384; i += 256) { int k = i >> 7, n = i & 127; Wt[n * LDA + k] = (_Float16)w2[i]; }
  __syncthreads();

  // layer 1 (K=4): VALU, write f16 to LDS
  for (int i = t; i < 16384; i += 256) {
    int r = i >> 7, c = i & 127;
    float a = b1s[c] + xs[r * 4 + 0] * w1s[c] + xs[r * 4 + 1] * w1s[128 + c]
                     + xs[r * 4 + 2] * w1s[256 + c] + xs[r * 4 + 3] * w1s[384 + c];
    Xh[r * LDA + c] = (_Float16)fmaxf(a, 0.f);
  }
  __syncthreads();

  // layer 2 via WMMA; only the per-column sum of relu(h2) is needed (mean pool)
  const int wave = t >> 5, lane = t & 31, half = lane >> 4, lm = lane & 15;
  for (int nt = 0; nt < 8; ++nt) {
    v8f acc = {};
#pragma unroll
    for (int kc = 0; kc < 4; ++kc) {
      int kb = kc * 32 + half * 8;
      v16h a = load_frag(Xh, wave * 16 + lm, kb);
      v16h b = load_frag(Wt, nt * 16 + lm, kb);
      acc = wmma16(a, b, acc);
    }
    int c = nt * 16 + lm;
    float bb = b2s[c];
    float s = 0.f;
#pragma unroll
    for (int v = 0; v < 8; ++v) s += fmaxf(acc[v] + bb, 0.f);
    s += __shfl_xor(s, 16);                 // combine the two half-tiles (rows 0-7 / 8-15)
    if (half == 0) wsum[wave * 128 + c] = s;
  }
  __syncthreads();
  // deterministic fixed-order reduction over the 8 waves
  for (int i = t; i < 128; i += 256) {
    float s = 0.f;
#pragma unroll
    for (int w = 0; w < 8; ++w) s += wsum[w * 128 + i];
    partial[blockIdx.x * 128 + i] = s;
  }
}

// ---------------- Kernel 2: pooled heads + serial EKF ODE (1 block) ----------------
__global__ __launch_bounds__(256) void ode_kernel(
    const float* __restrict__ partial,
    const float* __restrict__ poolw, const float* __restrict__ poolb,
    const float* __restrict__ muw,  const float* __restrict__ mub,
    const float* __restrict__ lvw,  const float* __restrict__ lvb,
    const float* __restrict__ lamw, const float* __restrict__ lamb,
    const float* __restrict__ w1,   const float* __restrict__ b1,   // [65,128]
    const float* __restrict__ w2,   const float* __restrict__ b2,   // [128,128]
    const float* __restrict__ w3,   const float* __restrict__ b3,   // [128,32]
    const float* __restrict__ tprev_p, const float* __restrict__ tnext_p,
    float* __restrict__ out, float* __restrict__ m_ws, float* __restrict__ C_ws) {
  __shared__ float pooled[128], pact[128], lvs[32];
  __shared__ float z[65], h1[128], r1[128], h2[128], r2[128];
  __shared__ float T[32 * 128], A[32 * 32], C[32 * 32], AC[32 * 32];
  __shared__ float m[32], fv[32], lam[32];
  const int t = threadIdx.x;

  // ODE weights are reused 20x; pull them toward the WGP while pooling runs
  prefetch_f32(w2, 16384, t);
  prefetch_f32(w1, 8320, t);
  prefetch_f32(w3, 4096, t);

  if (t < 128) {                       // mean over N (deterministic block order)
    float s = 0.f;
    for (int b = 0; b < 32; ++b) s += partial[b * 128 + t];
    pooled[t] = s * (1.f / 4096.f);
  }
  __syncthreads();
  if (t < 128) {                       // pooled = relu(pool(mean))
    float a = poolb[t];
    for (int k = 0; k < 128; ++k) a += pooled[k] * poolw[k * 128 + t];
    pact[t] = fmaxf(a, 0.f);
  }
  __syncthreads();
  if (t < 32) {                        // heads (these are outputs 3/4/5)
    float amu = mub[t], alv = lvb[t], ala = lamb[t];
    for (int k = 0; k < 128; ++k) {
      float p = pact[k];
      amu += p * muw[k * 32 + t]; alv += p * lvw[k * 32 + t]; ala += p * lamw[k * 32 + t];
    }
    out[16384 + t] = amu; out[16416 + t] = alv; out[16448 + t] = ala;
    m[t] = amu; lvs[t] = alv; lam[t] = ala;
  }
  __syncthreads();
  for (int i = t; i < 1024; i += 256) {            // C0 = diag(exp(logvar))
    int r = i >> 5, c = i & 31;
    C[i] = (r == c) ? expf(lvs[r]) : 0.f;
  }
  const float tprev = tprev_p[0], tnext = tnext_p[0];
  const float dt = (tnext - tprev) / (float)NSTEPS;
  float tcur = tprev;
  __syncthreads();

  for (int step = 0; step < NSTEPS; ++step) {
    if (t < 32) { z[t] = m[t]; z[32 + t] = lam[t]; }
    if (t == 64) z[64] = tcur;
    __syncthreads();
    if (t < 128) {
      float a = b1[t];
      for (int k = 0; k < 65; ++k) a += z[k] * w1[k * 128 + t];
      h1[t] = fmaxf(a, 0.f); r1[t] = a > 0.f ? 1.f : 0.f;
    }
    __syncthreads();
    if (t < 128) {
      float a = b2[t];
      for (int k = 0; k < 128; ++k) a += h1[k] * w2[k * 128 + t];
      h2[t] = fmaxf(a, 0.f); r2[t] = a > 0.f ? 1.f : 0.f;
    }
    __syncthreads();
    if (t < 32) {
      float a = b3[t];
      for (int k = 0; k < 128; ++k) a += h2[k] * w3[k * 32 + t];
      fv[t] = a;
    }
    __syncthreads();
    // T[i][j] = r2[j] * sum_k W1[i][k]*r1[k]*W2[k][j]   (i<32: phi rows of W1)
    for (int i = t; i < 4096; i += 256) {
      int r = i >> 7, j = i & 127;
      float s = 0.f;
      for (int k = 0; k < 128; ++k) s += w1[r * 128 + k] * r1[k] * w2[k * 128 + j];
      T[i] = r2[j] * s;
    }
    __syncthreads();
    // A = I + dt * (T @ W3)
    for (int i = t; i < 1024; i += 256) {
      int r = i >> 5, o = i & 31;
      float s = 0.f;
      for (int j = 0; j < 128; ++j) s += T[r * 128 + j] * w3[j * 32 + o];
      A[i] = ((r == o) ? 1.f : 0.f) + dt * s;
    }
    __syncthreads();
    for (int i = t; i < 1024; i += 256) {          // AC = A @ C
      int r = i >> 5, c = i & 31;
      float s = 0.f;
      for (int k = 0; k < 32; ++k) s += A[r * 32 + k] * C[k * 32 + c];
      AC[i] = s;
    }
    __syncthreads();
    for (int i = t; i < 1024; i += 256) {          // C = AC @ A^T + q*dt*I
      int r = i >> 5, c = i & 31;
      float s = 0.f;
      for (int k = 0; k < 32; ++k) s += AC[r * 32 + k] * A[c * 32 + k];
      C[i] = s + ((r == c) ? QNOISE * dt : 0.f);
    }
    if (t < 32) m[t] += dt * fv[t];
    tcur += dt;
    __syncthreads();
  }
  if (t < 32) m_ws[t] = m[t];
  for (int i = t; i < 1024; i += 256) C_ws[i] = C[i];
}

// ---------------- Kernel 3: modenet + diagonal-only covariance projection ----------------
__global__ __launch_bounds__(256) void mode_kernel(
    const float* __restrict__ coords,
    const float* __restrict__ w1, const float* __restrict__ b1,   // [2,128]
    const float* __restrict__ w2, const float* __restrict__ b2,   // [128,128]
    const float* __restrict__ w3, const float* __restrict__ b3,   // [128,32]
    const float* __restrict__ m_ws, const float* __restrict__ C_ws,
    float* __restrict__ out) {
  __shared__ alignas(16) _Float16 H1[128 * LDA];       // h1, then (in-place) h2
  __shared__ alignas(16) _Float16 Wt2[128 * LDA];      // mode.l2 W^T
  __shared__ alignas(16) _Float16 Wt3[32 * LDA];       // mode.l3 W^T
  __shared__ float Wrow[128][32];                      // final W rows (f32)
  __shared__ float Cs[1024], ms[32];
  __shared__ float w1s[256], b1s[128], b2s[128], b3s[32], xs[128 * 2];

  const int t = threadIdx.x;
  const int n0 = blockIdx.x * 128;

  prefetch_f32(w2, 16384, t);
  prefetch_f32(w3, 4096, t);

  for (int i = t; i < 256; i += 256) w1s[i] = w1[i];
  for (int i = t; i < 128; i += 256) { b1s[i] = b1[i]; b2s[i] = b2[i]; }
  for (int i = t; i < 32;  i += 256) b3s[i] = b3[i];
  for (int i = t; i < 1024;  i += 256) Cs[i] = C_ws[i];
  for (int i = t; i < 32;    i += 256) ms[i] = m_ws[i];
  for (int i = t; i < 128;   i += 256) { xs[2 * i] = coords[2 * (n0 + i)]; xs[2 * i + 1] = coords[2 * (n0 + i) + 1]; }
  for (int i = t; i < 16384; i += 256) { int k = i >> 7, n = i & 127; Wt2[n * LDA + k] = (_Float16)w2[i]; }
  for (int i = t; i < 4096;  i += 256) { int k = i >> 5, n = i & 31;  Wt3[n * LDA + k] = (_Float16)w3[i]; }
  __syncthreads();

  // layer 1 (K=2)
  for (int i = t; i < 16384; i += 256) {
    int r = i >> 7, c = i & 127;
    float a = b1s[c] + xs[2 * r] * w1s[c] + xs[2 * r + 1] * w1s[128 + c];
    H1[r * LDA + c] = (_Float16)fmaxf(a, 0.f);
  }
  __syncthreads();

  const int wave = t >> 5, lane = t & 31, half = lane >> 4, lm = lane & 15;

  // layer 2 WMMA: each wave owns rows [wave*16, wave*16+16) -> in-place update is hazard-free
  v8f acc2[8];
#pragma unroll
  for (int nt = 0; nt < 8; ++nt) {
    v8f acc = {};
#pragma unroll
    for (int kc = 0; kc < 4; ++kc) {
      int kb = kc * 32 + half * 8;
      v16h a = load_frag(H1, wave * 16 + lm, kb);
      v16h b = load_frag(Wt2, nt * 16 + lm, kb);
      acc = wmma16(a, b, acc);
    }
    acc2[nt] = acc;
  }
#pragma unroll
  for (int nt = 0; nt < 8; ++nt) {
    int col = nt * 16 + lm;
    float bb = b2s[col];
#pragma unroll
    for (int v = 0; v < 8; ++v) {
      int row = wave * 16 + v + 8 * half;          // D layout: M = v + 8*(lane>>4), N = lane&15
      H1[row * LDA + col] = (_Float16)fmaxf(acc2[nt][v] + bb, 0.f);
    }
  }
  __syncthreads();

  // layer 3 WMMA: [128,128] @ [128,32] -> stage rows in LDS (f32)
#pragma unroll
  for (int nt = 0; nt < 2; ++nt) {
    v8f acc = {};
#pragma unroll
    for (int kc = 0; kc < 4; ++kc) {
      int kb = kc * 32 + half * 8;
      v16h a = load_frag(H1, wave * 16 + lm, kb);
      v16h b = load_frag(Wt3, nt * 16 + lm, kb);
      acc = wmma16(a, b, acc);
    }
    int col = nt * 16 + lm;
    float bb = b3s[col];
#pragma unroll
    for (int v = 0; v < 8; ++v) {
      int row = wave * 16 + v + 8 * half;
      Wrow[row][col] = acc[v] + bb;
    }
  }
  __syncthreads();

  // coalesced W output (row-major [128,32] contiguous block per tile)
  for (int i = t; i < 4096; i += 256)
    out[16480 + (size_t)n0 * 32 + i] = Wrow[i >> 5][i & 31];

  // mu_u = Mv . m ; var_u = Mv C Mv^T  (diagonal only -- never materialize [2N,2N])
  {
    const int nl = t >> 1, j = t & 1;       // 128 rows x 2 complex components
    float Mv[32];
#pragma unroll
    for (int r = 0; r < 16; ++r) {
      float wr = Wrow[nl][2 * r], wi = Wrow[nl][2 * r + 1];
      Mv[2 * r]     = j ? wi : wr;
      Mv[2 * r + 1] = j ? wr : -wi;
    }
    float mu = 0.f;
#pragma unroll
    for (int k = 0; k < 32; ++k) mu += Mv[k] * ms[k];
    float var = 0.f;
#pragma unroll
    for (int i = 0; i < 32; ++i) {
      float s = 0.f;
#pragma unroll
      for (int k = 0; k < 32; ++k) s += Cs[i * 32 + k] * Mv[k];
      var += Mv[i] * s;
    }
    var = fmaxf(var, COVEPS);
    const int n = n0 + nl;
    out[2 * n + j] = mu;                 // mu_u
    out[8192 + 2 * n + j] = logf(var);   // logvar_u
  }
}

extern "C" void kernel_launch(void* const* d_in, const int* in_sizes, int n_in,
                              void* d_out, int out_size, void* d_ws, size_t ws_size,
                              hipStream_t stream) {
  const float* coords = (const float*)d_in[0];
  const float* yprev  = (const float*)d_in[1];
  const float* el1w = (const float*)d_in[2],  *el1b = (const float*)d_in[3];
  const float* el2w = (const float*)d_in[4],  *el2b = (const float*)d_in[5];
  const float* epw  = (const float*)d_in[6],  *epb  = (const float*)d_in[7];
  const float* emw  = (const float*)d_in[8],  *emb  = (const float*)d_in[9];
  const float* elvw = (const float*)d_in[10], *elvb = (const float*)d_in[11];
  const float* elaw = (const float*)d_in[12], *elab = (const float*)d_in[13];
  const float* o1w  = (const float*)d_in[14], *o1b  = (const float*)d_in[15];
  const float* o2w  = (const float*)d_in[16], *o2b  = (const float*)d_in[17];
  const float* o3w  = (const float*)d_in[18], *o3b  = (const float*)d_in[19];
  const float* m1w  = (const float*)d_in[20], *m1b  = (const float*)d_in[21];
  const float* m2w  = (const float*)d_in[22], *m2b  = (const float*)d_in[23];
  const float* m3w  = (const float*)d_in[24], *m3b  = (const float*)d_in[25];
  const float* tprev = (const float*)d_in[26];
  const float* tnext = (const float*)d_in[27];

  float* out = (float*)d_out;
  float* ws  = (float*)d_ws;
  float* partial = ws;            // 32*128
  float* m_ws    = ws + 4096;     // 32
  float* C_ws    = ws + 4128;     // 1024

  const int nblocks = in_sizes[0] / 256;   // N/128 row-tiles (N = in_sizes[0]/2)

  enc_kernel<<<nblocks, 256, 0, stream>>>(coords, yprev, el1w, el1b, el2w, el2b, partial);
  ode_kernel<<<1, 256, 0, stream>>>(partial, epw, epb, emw, emb, elvw, elvb, elaw, elab,
                                    o1w, o1b, o2w, o2b, o3w, o3b, tprev, tnext,
                                    out, m_ws, C_ws);
  mode_kernel<<<nblocks, 256, 0, stream>>>(coords, m1w, m1b, m2w, m2b, m3w, m3b,
                                           m_ws, C_ws, out);
}
